// MDL_RNN_mnist_53541062312355
// MI455X (gfx1250) — matrix operand order repaired and checked by reading, not compiled, not run on hardware
//
#include <hip/hip_runtime.h>
#include <hip/hip_bf16.h>
#include <math.h>

// ---------------------------------------------------------------------------
// Spiking low-rank RNN (MDL_RNN_mnist) for MI455X (gfx1250, wave32, WMMA).
//
// All matrix work uses V_WMMA_F32_16X16X32_F16 (f16 A/B, f32 accum):
//   * Wr = (l*pin) @ pout^T          (2048x2048, K=256) -> stored f16
//   * per step: I = spk_t @ Win^T + r @ Wr^T   (M=512, N=2048, K=800+2048)
// GEMM: 128x128 C-tile per 128-thread block (4 waves), 64x64 per wave
// (16 WMMA tiles, 128 acc VGPRs), double-buffered LDS with the CDNA5
// GLOBAL_LOAD_ASYNC_TO_LDS_B128 1-hop copy path (ASYNCcnt) when the
// toolchain exposes the builtin, else reg-staging fallback.
// ---------------------------------------------------------------------------

typedef _Float16 v16h __attribute__((ext_vector_type(16)));
typedef _Float16 v8h  __attribute__((ext_vector_type(8)));
typedef float    v8f  __attribute__((ext_vector_type(8)));
typedef int      ai4  __attribute__((ext_vector_type(4)));

#define T_STEPS 100
#define B_SZ    512
#define IN_SZ   784
#define INP     800      // IN padded to multiple of 32 (K-tile)
#define H_SZ    2048
#define P_SZ    256
#define OUT_SZ  10

#define LM   0.90483741803595957f   // exp(-1/10)
#define LD   0.96721610084254463f   // exp(-1/30)
#define LR   0.60653065971263342f   // exp(-1/2)
#define COEF (1.0f/60.0f)           // DT/(TAUD*TAUR)
#define VTHR_C 1.0f
#define TREF_C 2.0f
#define DT_C   1.0f

#define ROWP 40   // LDS row pitch in f16 (32 data + 8 pad) -> 80B, bank-skewed

#if defined(__has_builtin)
#if __has_builtin(__builtin_amdgcn_global_load_async_to_lds_b128) && \
    __has_builtin(__builtin_amdgcn_s_wait_asynccnt)
#define USE_ASYNC_LDS 1
#endif
#endif
#ifndef USE_ASYNC_LDS
#define USE_ASYNC_LDS 0
#endif

#if USE_ASYNC_LDS
// AMD clang-22 signature (from diagnostic): params are generic int4* typed
// pointers (global src, LDS dst, imm offset, imm cpol).
__device__ __forceinline__ void async_cp16(const _Float16* g, _Float16* l) {
  __builtin_amdgcn_global_load_async_to_lds_b128((ai4*)g, (ai4*)l, 0, 0);
}
#endif

// ---------------- counter-based RNG (Bernoulli rate coding) ----------------
__device__ __forceinline__ unsigned hash_u32(unsigned x) {
  x ^= x >> 16; x *= 0x7feb352dU;
  x ^= x >> 15; x *= 0x846ca68bU;
  x ^= x >> 16; return x;
}

// ---------------- elementwise prep kernels ----------------
__global__ void __launch_bounds__(256)
k_conv_pp(const float* __restrict__ pin, const float* __restrict__ pout,
          const float* __restrict__ l,
          _Float16* __restrict__ pinl16, _Float16* __restrict__ pout16) {
  int idx = blockIdx.x * 256 + threadIdx.x;          // < H*P
  int p = idx & (P_SZ - 1);
  pinl16[idx] = (_Float16)(l[p] * pin[idx]);
  pout16[idx] = (_Float16)pout[idx];
}

__global__ void __launch_bounds__(256)
k_conv_win(const float* __restrict__ Win, _Float16* __restrict__ Win16) {
  int idx = blockIdx.x * 256 + threadIdx.x;          // < H*INP
  int i = idx % INP, h = idx / INP;
  Win16[idx] = (i < IN_SZ) ? (_Float16)Win[h * IN_SZ + i] : (_Float16)0.0f;
}

__global__ void __launch_bounds__(256)
k_init_state(float* __restrict__ mem, float* __restrict__ s,
             float* __restrict__ hh, float* __restrict__ rr,
             float* __restrict__ tl, float* __restrict__ rm,
             _Float16* __restrict__ r16) {
  int idx = blockIdx.x * 256 + threadIdx.x;          // < B*H
  mem[idx] = 0.f; s[idx] = 0.f; hh[idx] = 0.f; rr[idx] = 0.f;
  tl[idx] = -1.f; rm[idx] = 0.f; r16[idx] = (_Float16)0.0f;
}

__global__ void __launch_bounds__(256)
k_spikes(const float* __restrict__ inputs, _Float16* __restrict__ spk16, int t) {
  int idx = blockIdx.x * 256 + threadIdx.x;          // < B*INP
  int i = idx % INP, b = idx / INP;
  float v = 0.f;
  if (i < IN_SZ) {
    float p = inputs[b * IN_SZ + i];
    unsigned c = (unsigned)idx + (unsigned)t * (B_SZ * INP) + 42u * 0x9E3779B9u;
    float u = (float)(hash_u32(c) >> 8) * (1.0f / 16777216.0f);
    v = (u < p) ? 1.0f : 0.0f;
  }
  spk16[idx] = (_Float16)v;
}

__global__ void __launch_bounds__(256)
k_update(const float* __restrict__ I,
         float* __restrict__ memb, float* __restrict__ sb,
         float* __restrict__ hb, float* __restrict__ rb,
         float* __restrict__ tlb, float* __restrict__ rmb,
         _Float16* __restrict__ r16, int t) {
  int idx = blockIdx.x * 256 + threadIdx.x;          // < B*H
  float time = DT_C * (float)t;
  float Iv  = I[idx];
  float mem = memb[idx], sv = sb[idx], hv = hb[idx];
  float rv  = rb[idx],  tl = tlb[idx], rmv = rmb[idx];

  float not_ref = (time > tl + TREF_C) ? 1.f : 0.f;
  mem = not_ref * (LM * mem + (1.f - LM) * Iv) * (1.f - sv);
  hv  = LR * hv + COEF * sv;                  // uses previous spikes
  rv  = LD * rv + DT_C * hv;
  float snew = (mem - VTHR_C > 0.f) ? 1.f : 0.f;
  tl  = tl + (time - tl) * snew;
  rmv = fmaxf(rmv, rv);

  memb[idx] = mem; sb[idx] = snew; hb[idx] = hv; rb[idx] = rv;
  tlb[idx] = tl; rmb[idx] = rmv;
  r16[idx] = (_Float16)rv;                    // f16 operand for next step GEMM
}

// ---------------- WMMA GEMM: C[m][n] = sum_k A[m][k]*B[n][k] --------------
// Two K-segments (A0/B0 then A1/B1) fuse the input-current and recurrent
// GEMMs. 128 threads (4 waves) per 128x128 C tile; each wave owns 64x64 =
// 4x4 WMMA tiles. K-tile = 32. LDS double buffer, row pitch 40 f16.
__global__ void __launch_bounds__(128)
gemm_dual(const _Float16* __restrict__ A0, int lda0,
          const _Float16* __restrict__ B0, int ldb0, int nk0,
          const _Float16* __restrict__ A1, int lda1,
          const _Float16* __restrict__ B1, int ldb1, int nk1,
          float* __restrict__ Cf, _Float16* __restrict__ Ch, int ldc) {
  __shared__ _Float16 ldsA[2][128 * ROWP];
  __shared__ _Float16 ldsB[2][128 * ROWP];

  const int tid   = threadIdx.x;
  const int lane  = tid & 31;
  const int wave  = tid >> 5;
  const int waveM = wave >> 1;            // 0..1
  const int waveN = wave & 1;             // 0..1
  const int m0 = blockIdx.y * 128;
  const int n0 = blockIdx.x * 128;
  const int nk = nk0 + nk1;

  // global->LDS copy mapping: thread t covers rows (t>>2)+32*i, 16B chunk
  // (t&3)*8 f16 -> 4 lanes cover one 64B row segment (coalesced per row).
  const int lrow = tid >> 2;              // 0..31
  const int lcol = (tid & 3) * 8;         // f16 units: 0,8,16,24

  v8f acc[4][4];
  v8f zero = {};
#pragma unroll
  for (int mi = 0; mi < 4; ++mi)
#pragma unroll
    for (int ni = 0; ni < 4; ++ni) acc[mi][ni] = zero;

  auto pick = [&](int kt, const _Float16*& A, const _Float16*& Bm,
                  int& lda, int& ldb, int& kl) {
    if (kt < nk0) { A = A0; Bm = B0; lda = lda0; ldb = ldb0; kl = kt * 32; }
    else          { A = A1; Bm = B1; lda = lda1; ldb = ldb1; kl = (kt - nk0) * 32; }
  };

#if USE_ASYNC_LDS
  auto issueTile = [&](int kt, int buf) {
    const _Float16 *A, *Bm; int lda, ldb, kl;
    pick(kt, A, Bm, lda, ldb, kl);
#pragma unroll
    for (int i = 0; i < 4; ++i) {
      int row = lrow + 32 * i;
      async_cp16(A  + (size_t)(m0 + row) * lda + kl + lcol,
                 &ldsA[buf][row * ROWP + lcol]);
      async_cp16(Bm + (size_t)(n0 + row) * ldb + kl + lcol,
                 &ldsB[buf][row * ROWP + lcol]);
    }
  };
#else
  uint4 ra[4], rb[4];
  auto loadG = [&](int kt) {
    const _Float16 *A, *Bm; int lda, ldb, kl;
    pick(kt, A, Bm, lda, ldb, kl);
#pragma unroll
    for (int i = 0; i < 4; ++i) {
      int row = lrow + 32 * i;
      ra[i] = *reinterpret_cast<const uint4*>(A  + (size_t)(m0 + row) * lda + kl + lcol);
      rb[i] = *reinterpret_cast<const uint4*>(Bm + (size_t)(n0 + row) * ldb + kl + lcol);
    }
  };
  auto storeL = [&](int buf) {
#pragma unroll
    for (int i = 0; i < 4; ++i) {
      int row = lrow + 32 * i;
      *reinterpret_cast<uint4*>(&ldsA[buf][row * ROWP + lcol]) = ra[i];
      *reinterpret_cast<uint4*>(&ldsB[buf][row * ROWP + lcol]) = rb[i];
    }
  };
#endif

  auto compute = [&](int buf) {
    v16h af[4], bf[4];
    // A fragment (16x32 f16): lane L -> row M=L%16; K-block base (L/16)*8,
    // VGPRs 0-3 = K base..base+7, VGPRs 4-7 = K base+16..base+23.
#pragma unroll
    for (int mi = 0; mi < 4; ++mi) {
      int rowA = waveM * 64 + mi * 16 + (lane & 15);
      int kb   = (lane >> 4) * 8;
      v8h lo = *reinterpret_cast<const v8h*>(&ldsA[buf][rowA * ROWP + kb]);
      v8h hi = *reinterpret_cast<const v8h*>(&ldsA[buf][rowA * ROWP + kb + 16]);
      af[mi] = __builtin_shufflevector(lo, hi, 0,1,2,3,4,5,6,7,8,9,10,11,12,13,14,15);
    }
    // B fragment (32x16 f16): lane L -> column N=L%16; K-block base (L/16)*8,
    // VGPRs 0-3 = K base..base+7, VGPRs 4-7 = K base+16..base+23. LDS tile is (N,K) row-major.
#pragma unroll
    for (int ni = 0; ni < 4; ++ni) {
      int colB = waveN * 64 + ni * 16 + (lane & 15);
      int kb   = (lane >> 4) * 8;
      v8h lo = *reinterpret_cast<const v8h*>(&ldsB[buf][colB * ROWP + kb]);
      v8h hi = *reinterpret_cast<const v8h*>(&ldsB[buf][colB * ROWP + kb + 16]);
      bf[ni] = __builtin_shufflevector(lo, hi, 0,1,2,3,4,5,6,7,8,9,10,11,12,13,14,15);
    }
#pragma unroll
    for (int mi = 0; mi < 4; ++mi)
#pragma unroll
      for (int ni = 0; ni < 4; ++ni)
        acc[mi][ni] = __builtin_amdgcn_wmma_f32_16x16x32_f16(
            false, af[mi], false, bf[ni], (short)0, acc[mi][ni], false, false);
  };

#if USE_ASYNC_LDS
  issueTile(0, 0);
  __builtin_amdgcn_s_wait_asynccnt(0);
  __syncthreads();
  for (int kt = 0; kt < nk; ++kt) {
    if (kt + 1 < nk) issueTile(kt + 1, (kt + 1) & 1);   // writes other buffer
    compute(kt & 1);
    if (kt + 1 < nk) {
      __builtin_amdgcn_s_wait_asynccnt(0);              // own async loads done
      __syncthreads();                                  // -> all waves' done
    }
  }
#else
  loadG(0);
  storeL(0);
  __syncthreads();
  for (int kt = 0; kt < nk; ++kt) {
    if (kt + 1 < nk) loadG(kt + 1);
    compute(kt & 1);
    if (kt + 1 < nk) {
      __syncthreads();
      storeL((kt + 1) & 1);
      __syncthreads();
    }
  }
#endif

  // C/D layout: VGPR v on lane L -> (M = (L/16)*8 + v, N = L%16)
#pragma unroll
  for (int mi = 0; mi < 4; ++mi)
#pragma unroll
    for (int ni = 0; ni < 4; ++ni) {
      int r0 = m0 + waveM * 64 + mi * 16 + (lane >> 4) * 8;
      int c0 = n0 + waveN * 64 + ni * 16 + (lane & 15);
      if (Ch) {
#pragma unroll
        for (int v = 0; v < 8; ++v) Ch[(size_t)(r0 + v) * ldc + c0] = (_Float16)acc[mi][ni][v];
      } else {
#pragma unroll
        for (int v = 0; v < 8; ++v) Cf[(size_t)(r0 + v) * ldc + c0] = acc[mi][ni][v];
      }
    }
}

// ---------------- readout + softmax (tiny: 512x10, K=2048) ----------------
__global__ void __launch_bounds__(256)
k_head_softmax(const float* __restrict__ rm, const float* __restrict__ Wout,
               float* __restrict__ out) {
  __shared__ float red[256];
  __shared__ float outs[OUT_SZ];
  int b = blockIdx.x, tid = threadIdx.x;
  float part[OUT_SZ];
#pragma unroll
  for (int o = 0; o < OUT_SZ; ++o) part[o] = 0.f;
  for (int h = tid; h < H_SZ; h += 256) {
    float r = rm[(size_t)b * H_SZ + h];
#pragma unroll
    for (int o = 0; o < OUT_SZ; ++o) part[o] += r * Wout[(size_t)o * H_SZ + h];
  }
  for (int o = 0; o < OUT_SZ; ++o) {
    red[tid] = part[o];
    __syncthreads();
    for (int s = 128; s > 0; s >>= 1) {
      if (tid < s) red[tid] += red[tid + s];
      __syncthreads();
    }
    if (tid == 0) outs[o] = red[0];
    __syncthreads();
  }
  if (tid == 0) {
    float mx = -1e30f;
    for (int o = 0; o < OUT_SZ; ++o) mx = fmaxf(mx, outs[o]);
    float sum = 0.f, e[OUT_SZ];
    for (int o = 0; o < OUT_SZ; ++o) { e[o] = expf(outs[o] - mx); sum += e[o]; }
    float inv = 1.f / sum;
    for (int o = 0; o < OUT_SZ; ++o) out[(size_t)b * OUT_SZ + o] = e[o] * inv;
  }
}

// ---------------------------------------------------------------------------
extern "C" void kernel_launch(void* const* d_in, const int* in_sizes, int n_in,
                              void* d_out, int out_size, void* d_ws, size_t ws_size,
                              hipStream_t stream) {
  (void)in_sizes; (void)n_in; (void)out_size; (void)ws_size;
  const float* inputs = (const float*)d_in[0];   // (B, IN)
  const float* Win    = (const float*)d_in[1];   // (H, IN)
  const float* pin    = (const float*)d_in[2];   // (H, P)
  const float* pout   = (const float*)d_in[3];   // (H, P)
  const float* l      = (const float*)d_in[4];   // (P,)
  const float* Wout   = (const float*)d_in[5];   // (OUT, H)
  float* out = (float*)d_out;                    // (B, OUT)

  char* ws = (char*)d_ws;
  size_t off = 0;
  auto alloc = [&](size_t bytes) { size_t o = off; off = (off + bytes + 255) & ~(size_t)255; return o; };
  _Float16* Wr16   = (_Float16*)(ws + alloc((size_t)H_SZ * H_SZ * 2));
  _Float16* Win16  = (_Float16*)(ws + alloc((size_t)H_SZ * INP * 2));
  _Float16* pinl16 = (_Float16*)(ws + alloc((size_t)H_SZ * P_SZ * 2));
  _Float16* pout16 = (_Float16*)(ws + alloc((size_t)H_SZ * P_SZ * 2));
  _Float16* spk16  = (_Float16*)(ws + alloc((size_t)B_SZ * INP * 2));
  _Float16* r16    = (_Float16*)(ws + alloc((size_t)B_SZ * H_SZ * 2));
  float* Ibuf = (float*)(ws + alloc((size_t)B_SZ * H_SZ * 4));
  float* memb = (float*)(ws + alloc((size_t)B_SZ * H_SZ * 4));
  float* sb   = (float*)(ws + alloc((size_t)B_SZ * H_SZ * 4));
  float* hb   = (float*)(ws + alloc((size_t)B_SZ * H_SZ * 4));
  float* rb   = (float*)(ws + alloc((size_t)B_SZ * H_SZ * 4));
  float* tlb  = (float*)(ws + alloc((size_t)B_SZ * H_SZ * 4));
  float* rmb  = (float*)(ws + alloc((size_t)B_SZ * H_SZ * 4));

  dim3 blk256(256), blk128(128);

  // ---- prep: f16 operands + Wr = (l*pin) @ pout^T (f16, 2048x2048) ----
  k_conv_pp <<<(H_SZ * P_SZ) / 256, blk256, 0, stream>>>(pin, pout, l, pinl16, pout16);
  k_conv_win<<<(H_SZ * INP)  / 256, blk256, 0, stream>>>(Win, Win16);
  gemm_dual<<<dim3(H_SZ / 128, H_SZ / 128), blk128, 0, stream>>>(
      pinl16, P_SZ, pout16, P_SZ, P_SZ / 32,
      nullptr, 0, nullptr, 0, 0,
      nullptr, Wr16, H_SZ);
  k_init_state<<<(B_SZ * H_SZ) / 256, blk256, 0, stream>>>(memb, sb, hb, rb, tlb, rmb, r16);

  // ---- T serially-dependent steps ----
  for (int t = 0; t < T_STEPS; ++t) {
    k_spikes<<<(B_SZ * INP) / 256, blk256, 0, stream>>>(inputs, spk16, t);
    // I = spk_t @ Win^T + r @ Wr^T   (M=512, N=2048, K = 800 + 2048)
    gemm_dual<<<dim3(H_SZ / 128, B_SZ / 128), blk128, 0, stream>>>(
        spk16, INP, Win16, INP, INP / 32,
        r16, H_SZ, Wr16, H_SZ, H_SZ / 32,
        Ibuf, nullptr, H_SZ);
    k_update<<<(B_SZ * H_SZ) / 256, blk256, 0, stream>>>(Ibuf, memb, sb, hb, rb, tlb, rmb, r16, t);
  }

  // ---- readout: softmax(rm @ Wout^T) ----
  k_head_softmax<<<B_SZ, blk256, 0, stream>>>(rmb, Wout, out);
}
